// DifferentialAttention_20194936226100
// MI455X (gfx1250) — matrix operand order repaired and checked by reading, not compiled, not run on hardware
//
#include <hip/hip_runtime.h>
#include <hip/hip_bf16.h>

// ---------------- problem constants ----------------
#define DIMN      2048
#define NHEADS    16
#define HEADDIM   64          // dh
#define SEQLEN    2048
#define BATCH     2
#define LAMBDA_INIT_F 0.7999593627f   // 0.8 - 0.6*exp(-0.3*32)

typedef __attribute__((ext_vector_type(16))) __bf16 v16bf;
typedef __attribute__((ext_vector_type(8)))  float  v8f;

static __device__ __forceinline__ v8f wmma_bf16_f32(v16bf a, v16bf b, v8f c) {
  // D = A(16x32 bf16) * B(32x16 bf16) + C(16x16 f32)
  return __builtin_amdgcn_wmma_f32_16x16x32_bf16(
      /*neg_a=*/false, a, /*neg_b=*/false, b,
      /*c_mod=*/(short)0, c, /*reuse_a=*/false, /*reuse_b=*/false);
}

// ---------------- async global->LDS (CDNA5 ASYNCcnt path) ----------------
#if __has_builtin(__builtin_amdgcn_global_load_async_to_lds_b128)
#define HAVE_ASYNC 1
#else
#define HAVE_ASYNC 0
#endif

// builtin expects: (int __vector_size(16) addrspace(1)*, int __vector_size(16) addrspace(3)*, imm, imm)
typedef int v4i_vs __attribute__((vector_size(16)));
typedef __attribute__((address_space(1))) v4i_vs GV4;   // global int4
typedef __attribute__((address_space(3))) v4i_vs LV4;   // LDS int4

static __device__ __forceinline__ void async_copy16(const void* g, void* l) {
#if HAVE_ASYNC
  __builtin_amdgcn_global_load_async_to_lds_b128((GV4*)g, (LV4*)l, 0, 0);
#else
  *(uint4*)l = *(const uint4*)g;
#endif
}

static __device__ __forceinline__ void async_wait0() {
#if HAVE_ASYNC
#if __has_builtin(__builtin_amdgcn_s_wait_asynccnt)
  __builtin_amdgcn_s_wait_asynccnt(0);
#else
  asm volatile("s_wait_asynccnt 0x0" ::: "memory");
#endif
#endif
}

// ---------------- DPP 16-lane row reductions (no LDS round trip) ----------
template <int CTRL>
static __device__ __forceinline__ float dppmov(float x) {
  return __builtin_bit_cast(
      float, __builtin_amdgcn_update_dpp(0, __builtin_bit_cast(int, x), CTRL,
                                         0xf, 0xf, true));
}
static __device__ __forceinline__ float rmax16(float x) {
  x = fmaxf(x, dppmov<0x128>(x));   // row_ror:8
  x = fmaxf(x, dppmov<0x124>(x));   // row_ror:4
  x = fmaxf(x, dppmov<0x122>(x));   // row_ror:2
  x = fmaxf(x, dppmov<0x121>(x));   // row_ror:1
  return x;
}
static __device__ __forceinline__ float rsum16(float x) {
  x += dppmov<0x128>(x);
  x += dppmov<0x124>(x);
  x += dppmov<0x122>(x);
  x += dppmov<0x121>(x);
  return x;
}

// ---------------- fp32 -> bf16 conversion ----------------
__global__ __launch_bounds__(256)
void cvt_bf16_kernel(const float* __restrict__ src, __bf16* __restrict__ dst, int n) {
  int i4 = (blockIdx.x * blockDim.x + threadIdx.x) * 4;
  if (i4 + 3 < n) {
    float4 f = *(const float4*)(src + i4);
    dst[i4 + 0] = (__bf16)f.x;
    dst[i4 + 1] = (__bf16)f.y;
    dst[i4 + 2] = (__bf16)f.z;
    dst[i4 + 3] = (__bf16)f.w;
  }
}

// ---------------- bf16 WMMA GEMM:  C(MxN) = A(MxK) * W(NxK)^T -------------
// 256 threads (8 waves), 128x128 tile, K-step 32, double-buffered async LDS.
// wave w: M-rows [(w>>1)*32, +32), N-cols [(w&1)*64, +64) of the tile.
template <typename OutT>
__global__ __launch_bounds__(256)
void gemm_bf16_wmma(const __bf16* __restrict__ A, const __bf16* __restrict__ W,
                    OutT* __restrict__ C, int M, int N, int K) {
  __shared__ __bf16 sA[2][128 * 32];
  __shared__ __bf16 sW[2][128 * 32];
  const int tid  = threadIdx.x;
  const int wave = tid >> 5;
  const int lane = tid & 31;
  const int m0 = blockIdx.y * 128;
  const int n0 = blockIdx.x * 128;
  const int mg = (wave >> 1) * 32;
  const int ng = (wave & 1) * 64;

  const int ldRow = tid >> 1;          // 0..127
  const int ldOff = (tid & 1) * 16;    // 0 or 16 (elems)

  v8f acc[2][4] = {};

  auto stage = [&](int buf, int k0) {
    const __bf16* gA = &A[(size_t)(m0 + ldRow) * K + k0 + ldOff];
    const __bf16* gW = &W[(size_t)(n0 + ldRow) * K + k0 + ldOff];
    __bf16* lA = &sA[buf][ldRow * 32 + ldOff];
    __bf16* lW = &sW[buf][ldRow * 32 + ldOff];
    async_copy16(gA,     lA);
    async_copy16(gA + 8, lA + 8);
    async_copy16(gW,     lW);
    async_copy16(gW + 8, lW + 8);
  };

  stage(0, 0);
  async_wait0();
  __syncthreads();

  const int ar  = lane & 15;
  const int ach = (lane >> 4) * 16;

  for (int k0 = 0; k0 < K; k0 += 32) {
    const int buf = (k0 >> 5) & 1;
    if (k0 + 32 < K) stage(buf ^ 1, k0 + 32);             // overlap with WMMA
    __builtin_prefetch(&A[(size_t)(m0 + ldRow) * K + k0 + 64 + ldOff], 0, 1);

    v16bf aF0 = *(const v16bf*)&sA[buf][(mg + ar) * 32 + ach];
    v16bf aF1 = *(const v16bf*)&sA[buf][(mg + 16 + ar) * 32 + ach];
#pragma unroll
    for (int t = 0; t < 4; ++t) {
      v16bf bF = *(const v16bf*)&sW[buf][(ng + t * 16 + ar) * 32 + ach];
      acc[0][t] = wmma_bf16_f32(aF0, bF, acc[0][t]);
      acc[1][t] = wmma_bf16_f32(aF1, bF, acc[1][t]);
    }
    async_wait0();
    __syncthreads();
  }

  const int cn = lane & 15;
  const int rb = (lane < 16) ? 0 : 8;
#pragma unroll
  for (int i = 0; i < 2; ++i)
#pragma unroll
    for (int t = 0; t < 4; ++t)
#pragma unroll
      for (int r = 0; r < 8; ++r)
        C[(size_t)(m0 + mg + i * 16 + rb + r) * N + (n0 + ng + t * 16 + cn)] =
            (OutT)acc[i][t][r];
}

// ---------------- RoPE (in-place, bf16 (B,S,32,64)), with output scale ----
__global__ __launch_bounds__(256)
void rope_kernel(__bf16* __restrict__ t, const float* __restrict__ cosb,
                 const float* __restrict__ sinb, float oscale) {
  int pid = blockIdx.x * blockDim.x + threadIdx.x;       // pair id
  int p    = pid & 31;
  int head = (pid >> 5) & 31;
  int s    = (pid >> 10) & (SEQLEN - 1);
  int b    = pid >> 21;
  if (b >= BATCH) return;
  size_t idx = (((size_t)b * SEQLEN + s) * 32 + head) * 64 + 2 * p;
  float a  = (float)t[idx];
  float bb = (float)t[idx + 1];
  float c  = cosb[s * 32 + p];
  float sn = sinb[s * 32 + p];
  t[idx]     = (__bf16)((a * c - bb * sn) * oscale);
  t[idx + 1] = (__bf16)((a * sn + bb * c) * oscale);
}

// ---------------- lambda scalar ----------------
__global__ void lambda_kernel(const float* lq1, const float* lk1,
                              const float* lq2, const float* lk2,
                              float* lam_out) {
  __shared__ float s1[64], s2[64];
  int tn = threadIdx.x;
  s1[tn] = lq1[tn] * lk1[tn];
  s2[tn] = lq2[tn] * lk2[tn];
  __syncthreads();
  if (tn == 0) {
    float a = 0.f, b = 0.f;
    for (int i = 0; i < 64; ++i) { a += s1[i]; b += s2[i]; }
    *lam_out = __expf(a) - __expf(b) + LAMBDA_INIT_F;
  }
}

// ---------------- flash differential attention ----------------
// grid: (SEQ/64, NHEADS, BATCH); block 128 (4 waves, 16 q-rows each).
// Q was pre-scaled by dh^-0.5 in rope_kernel.
__global__ __launch_bounds__(128)
void diff_attn_kernel(const __bf16* __restrict__ q, const __bf16* __restrict__ k,
                      const __bf16* __restrict__ v, float* __restrict__ out,
                      int variant) {
  __shared__ __bf16 sK[32 * 64];        // 32 k-rows x dh
  __shared__ __bf16 sV[32 * 128];       // 32 k-rows x 2*dh
  __shared__ __bf16 sP[4][16 * 32];     // per-wave P staging

  const int tid  = threadIdx.x;
  const int wave = tid >> 5;
  const int lane = tid & 31;
  const int qblk = blockIdx.x;
  const int h    = blockIdx.y;
  const int b    = blockIdx.z;
  const int qh   = 2 * h + variant;

  const __bf16* qBase = q + (size_t)b * SEQLEN * DIMN + (size_t)qh * HEADDIM;
  const __bf16* kBase = k + (size_t)b * SEQLEN * DIMN + (size_t)qh * HEADDIM;
  const __bf16* vBase = v + (size_t)b * SEQLEN * DIMN + (size_t)h * 128;

  const int qr0 = qblk * 64 + wave * 16;
  const int ar  = lane & 15;
  const int ach = (lane >> 4) * 16;

  v16bf aQ0 = *(const v16bf*)&qBase[(size_t)(qr0 + ar) * DIMN + ach];
  v16bf aQ1 = *(const v16bf*)&qBase[(size_t)(qr0 + ar) * DIMN + 32 + ach];

  v8f acc[8] = {};
  float mrow[8], lrow[8];
#pragma unroll
  for (int r = 0; r < 8; ++r) { mrow[r] = -3.0e38f; lrow[r] = 0.f; }

  const int colLane = lane & 15;
  const int rowOff  = (lane < 16) ? 0 : 8;
  const int nkt     = qblk * 2 + 2;     // causal bound, 32-wide k tiles

  const int krow = tid >> 2;            // 0..31
  const int kq   = (tid & 3) * 16;
  const int vq   = (tid & 3) * 32;

  for (int kt = 0; kt < nkt; ++kt) {
    const int kc0 = kt * 32;
    // async stage of K-tile (32x64) and V-tile (32x128)
    {
      const __bf16* gK = &kBase[(size_t)(kc0 + krow) * DIMN + kq];
      const __bf16* gV = &vBase[(size_t)(kc0 + krow) * DIMN + vq];
      async_copy16(gK,      &sK[krow * 64 + kq]);
      async_copy16(gK + 8,  &sK[krow * 64 + kq + 8]);
      async_copy16(gV,      &sV[krow * 128 + vq]);
      async_copy16(gV + 8,  &sV[krow * 128 + vq + 8]);
      async_copy16(gV + 16, &sV[krow * 128 + vq + 16]);
      async_copy16(gV + 24, &sV[krow * 128 + vq + 24]);
    }
    async_wait0();
    __syncthreads();

    // scores: S(16x32) = Q(16x64) @ K^T
    v8f st[2];
#pragma unroll
    for (int j = 0; j < 2; ++j) {
      v8f s = {};
      v16bf b0 = *(const v16bf*)&sK[(j * 16 + ar) * 64 + ach];
      v16bf b1 = *(const v16bf*)&sK[(j * 16 + ar) * 64 + 32 + ach];
      s = wmma_bf16_f32(aQ0, b0, s);
      s = wmma_bf16_f32(aQ1, b1, s);
      st[j] = s;
    }

    // online softmax (DPP row reductions)
#pragma unroll
    for (int r = 0; r < 8; ++r) {
      int rowG = qr0 + rowOff + r;
      float s0 = st[0][r];
      float s1 = st[1][r];
      if (kc0 + colLane > rowG)      s0 = -3.0e38f;
      if (kc0 + 16 + colLane > rowG) s1 = -3.0e38f;
      float mx    = rmax16(fmaxf(s0, s1));
      float mnew  = fmaxf(mrow[r], mx);
      float alpha = __expf(mrow[r] - mnew);
      float p0 = __expf(s0 - mnew);
      float p1 = __expf(s1 - mnew);
      float ps = rsum16(p0 + p1);
      lrow[r] = lrow[r] * alpha + ps;
      mrow[r] = mnew;
#pragma unroll
      for (int t2 = 0; t2 < 8; ++t2) acc[t2][r] *= alpha;
      sP[wave][(rowOff + r) * 32 + colLane]      = (__bf16)p0;
      sP[wave][(rowOff + r) * 32 + 16 + colLane] = (__bf16)p1;
    }

    // O += P(16x32) @ V(32x128)
    v16bf pA = *(const v16bf*)&sP[wave][ar * 32 + ach];
#pragma unroll
    for (int t2 = 0; t2 < 8; ++t2) {
      v16bf vF;
#pragma unroll
      for (int s = 0; s < 16; ++s) vF[s] = sV[(ach + s) * 128 + t2 * 16 + ar];
      acc[t2] = wmma_bf16_f32(pA, vF, acc[t2]);
    }
    __syncthreads();
  }

  // epilogue: normalize by row sum, scatter fp32
#pragma unroll
  for (int r = 0; r < 8; ++r) {
    float il = 1.0f / lrow[r];
    int rowG = qr0 + rowOff + r;
#pragma unroll
    for (int t2 = 0; t2 < 8; ++t2)
      out[((size_t)b * SEQLEN + rowG) * DIMN + h * 128 + t2 * 16 + colLane] =
          acc[t2][r] * il;
  }
}

// ---------------- differential combine + per-head RMSNorm ----------------
__global__ __launch_bounds__(128)
void combine_kernel(const float* __restrict__ a1, const float* __restrict__ a2,
                    const float* __restrict__ lam_p, const float* __restrict__ subln,
                    __bf16* __restrict__ outb) {
  const float lam = *lam_p;
  size_t base = (size_t)blockIdx.x * 128;
  int tn = threadIdx.x;
  float d = a1[base + tn] - lam * a2[base + tn];
  float sq = rsum16(d * d);
  sq += __shfl_xor(sq, 16, 32);   // fold the two DPP rows of the wave
  __shared__ float red[4];
  if ((tn & 31) == 0) red[tn >> 5] = sq;
  __syncthreads();
  float tot = red[0] + red[1] + red[2] + red[3];
  float rms = rsqrtf(tot * (1.0f / 128.0f) + 1e-5f);
  float o = d * rms * subln[tn] * (1.0f - LAMBDA_INIT_F);
  outb[base + tn] = (__bf16)o;
}

// ---------------- host-side orchestration ----------------
extern "C" void kernel_launch(void* const* d_in, const int* in_sizes, int n_in,
                              void* d_out, int out_size, void* d_ws, size_t ws_size,
                              hipStream_t stream) {
  const float* x    = (const float*)d_in[0];
  const float* wq   = (const float*)d_in[1];
  const float* wk   = (const float*)d_in[2];
  const float* wv   = (const float*)d_in[3];
  const float* wo   = (const float*)d_in[4];
  const float* lq1  = (const float*)d_in[5];
  const float* lk1  = (const float*)d_in[6];
  const float* lq2  = (const float*)d_in[7];
  const float* lk2  = (const float*)d_in[8];
  const float* subln= (const float*)d_in[9];
  const float* rc   = (const float*)d_in[10];
  const float* rs   = (const float*)d_in[11];
  float* out = (float*)d_out;

  const size_t NX = (size_t)BATCH * SEQLEN * DIMN;   // 8,388,608
  const size_t NW = (size_t)DIMN * DIMN;             // 4,194,304

  __bf16* xb  = (__bf16*)d_ws;
  __bf16* wqb = xb  + NX;
  __bf16* wkb = wqb + NW;
  __bf16* wvb = wkb + NW;
  __bf16* wob = wvb + NW;
  __bf16* qb  = wob + NW;
  __bf16* kb  = qb  + NX;
  __bf16* vb  = kb  + NX;
  __bf16* cat = vb  + NX;
  float*  a1  = (float*)(cat + NX);
  float*  a2  = a1 + NX;
  float*  lam = a2 + NX;

  const int M = BATCH * SEQLEN;   // 4096

  cvt_bf16_kernel<<<(int)(NX / 4 + 255) / 256, 256, 0, stream>>>(x,  xb,  (int)NX);
  cvt_bf16_kernel<<<(int)(NW / 4 + 255) / 256, 256, 0, stream>>>(wq, wqb, (int)NW);
  cvt_bf16_kernel<<<(int)(NW / 4 + 255) / 256, 256, 0, stream>>>(wk, wkb, (int)NW);
  cvt_bf16_kernel<<<(int)(NW / 4 + 255) / 256, 256, 0, stream>>>(wv, wvb, (int)NW);
  cvt_bf16_kernel<<<(int)(NW / 4 + 255) / 256, 256, 0, stream>>>(wo, wob, (int)NW);

  dim3 ggrid(DIMN / 128, M / 128);
  gemm_bf16_wmma<__bf16><<<ggrid, 256, 0, stream>>>(xb, wqb, qb, M, DIMN, DIMN);
  gemm_bf16_wmma<__bf16><<<ggrid, 256, 0, stream>>>(xb, wkb, kb, M, DIMN, DIMN);
  gemm_bf16_wmma<__bf16><<<ggrid, 256, 0, stream>>>(xb, wvb, vb, M, DIMN, DIMN);

  const int npairs = (int)(NX / 2);
  rope_kernel<<<(npairs + 255) / 256, 256, 0, stream>>>(qb, rc, rs, 0.125f);
  rope_kernel<<<(npairs + 255) / 256, 256, 0, stream>>>(kb, rc, rs, 1.0f);

  lambda_kernel<<<1, 64, 0, stream>>>(lq1, lk1, lq2, lk2, lam);

  dim3 agrid(SEQLEN / 64, NHEADS, BATCH);
  diff_attn_kernel<<<agrid, 128, 0, stream>>>(qb, kb, vb, a1, 0);
  diff_attn_kernel<<<agrid, 128, 0, stream>>>(qb, kb, vb, a2, 1);

  combine_kernel<<<BATCH * SEQLEN * NHEADS, 128, 0, stream>>>(a1, a2, lam, subln, cat);

  gemm_bf16_wmma<float><<<ggrid, 256, 0, stream>>>(cat, wob, out, M, DIMN, DIMN);
}
